// SyntheticNeurotransmitterSystem_10763188044130
// MI455X (gfx1250) — matrix operand order repaired and checked by reading, not compiled, not run on hardware
//
#include <hip/hip_runtime.h>
#include <math.h>

// ---------------------------------------------------------------------------
// SyntheticNeurotransmitterSystem for MI455X (gfx1250, wave32, WMMA).
// Pipeline of fused kernels; all dense algebra via v_wmma_f32_16x16x32_f16.
// Workspace requirement: ~132 MB (4MB header + 5 x [T,256] f32 + [T,768] f32).
// ---------------------------------------------------------------------------

#define TOK   16384      // 8 * 2048 tokens
#define SEQ   2048
#define NB    8
#define ED    256

typedef __attribute__((ext_vector_type(16))) _Float16 v16h;
typedef __attribute__((ext_vector_type(8)))  _Float16 v8h;
typedef __attribute__((ext_vector_type(8)))  float    v8f;

// Load 16 contiguous f16 (32B) as two 16B vectors (16B alignment only needed).
__device__ __forceinline__ v16h ld16h(const _Float16* p) {
  const v8h* q = (const v8h*)p;
  v8h lo = q[0];
  v8h hi = q[1];
  v16h r;
#pragma unroll
  for (int j = 0; j < 8; ++j) { r[j] = lo[j]; r[j + 8] = hi[j]; }
  return r;
}

__device__ __forceinline__ unsigned pack2h(float a, float b) {
  union { _Float16 h[2]; unsigned u; } r;
  r.h[0] = (_Float16)a; r.h[1] = (_Float16)b;
  return r.u;
}

// ---- DPP16 butterfly reductions across a 16-lane half (no LDS traffic) ----
template<int CTRL>
__device__ __forceinline__ float dppf(float v) {
  return __int_as_float(
      __builtin_amdgcn_update_dpp(0, __float_as_int(v), CTRL, 0xF, 0xF, true));
}
__device__ __forceinline__ float half16_max(float v) {
  v = fmaxf(v, dppf<0xB1>(v));    // quad_perm [1,0,3,2]  : xor 1
  v = fmaxf(v, dppf<0x4E>(v));    // quad_perm [2,3,0,1]  : xor 2
  v = fmaxf(v, dppf<0x141>(v));   // row_half_mirror      : xor 7 (== xor4 here)
  v = fmaxf(v, dppf<0x128>(v));   // row_ror:8            : xor 8 within 16-row
  return v;
}
__device__ __forceinline__ float half16_sum(float v) {
  v += dppf<0xB1>(v);
  v += dppf<0x4E>(v);
  v += dppf<0x141>(v);
  v += dppf<0x128>(v);
  return v;
}

template<int ACT> __device__ __forceinline__ float actf(float v) {
  if constexpr (ACT == 1) return fmaxf(v, 0.0f);                                   // relu
  else if constexpr (ACT == 2) return 1.0f / (1.0f + __expf(-v));                  // sigmoid
  else if constexpr (ACT == 3) return tanhf(v);                                    // tanh
  else if constexpr (ACT == 4) return 0.5f * v * (1.0f + erff(v * 0.7071067811865475f)); // exact gelu
  else return v;
}

__device__ __forceinline__ float wave_sum(float v) {
  v += __shfl_xor(v, 16); v += __shfl_xor(v, 8); v += __shfl_xor(v, 4);
  v += __shfl_xor(v, 2);  v += __shfl_xor(v, 1);
  return v;
}

// ---------------------------------------------------------------------------
// Weight conversion fp32 -> f16
// ---------------------------------------------------------------------------
__global__ void cvt_f16_kernel(const float* __restrict__ s, _Float16* __restrict__ d, int n) {
  int i = blockIdx.x * 256 + threadIdx.x;
  if (i < n) d[i] = (_Float16)s[i];
}

// ---------------------------------------------------------------------------
// Generic GEMM: Out[T,N] = act( X[T,256] @ W^T + bias ), W f16 [N,256] row-major.
// One wave = 16(M) x 64(N) tile, K chunks of 32 -> v_wmma_f32_16x16x32_f16.
// ---------------------------------------------------------------------------
template<int ACT>
__global__ __launch_bounds__(256) void gemm_kernel(
    const float* __restrict__ X, const _Float16* __restrict__ W,
    const float* __restrict__ bias, float* __restrict__ Out, int N)
{
  const int K = ED;
  int wave = blockIdx.x * 8 + (threadIdx.x >> 5);
  int lane = threadIdx.x & 31;
  int nt64 = N >> 6;
  int mt = wave / nt64;
  int nt = wave - mt * nt64;
  int row0 = mt << 4;
  int n0   = nt << 6;
  if (row0 >= TOK) return;
  int m  = lane & 15;
  int hs = lane >> 4;     // 0: lanes 0-15, 1: lanes 16-31
  int o  = hs << 3;       // K-octet select per ISA A-fragment layout

  v8f acc[4];
#pragma unroll
  for (int t = 0; t < 4; ++t)
#pragma unroll
    for (int r = 0; r < 8; ++r) acc[t][r] = 0.0f;

  const float* xr = X + (size_t)(row0 + m) * K;
#pragma unroll
  for (int kc = 0; kc < K; kc += 32) {
    if (kc + 32 < K) __builtin_prefetch(xr + kc + 32, 0, 0);
    // A fragment (16x32 f16): lanes<16 hold K {0..15}, lanes>=16 {16..31}
    float4 f0 = *(const float4*)(xr + kc + (hs << 4));
    float4 f1 = *(const float4*)(xr + kc + (hs << 4) + 4);
    float4 f2 = *(const float4*)(xr + kc + (hs << 4) + 8);
    float4 f3 = *(const float4*)(xr + kc + (hs << 4) + 12);
    v16h a;
    a[0]=(_Float16)f0.x;  a[1]=(_Float16)f0.y;  a[2]=(_Float16)f0.z;  a[3]=(_Float16)f0.w;
    a[4]=(_Float16)f1.x;  a[5]=(_Float16)f1.y;  a[6]=(_Float16)f1.z;  a[7]=(_Float16)f1.w;
    a[8]=(_Float16)f2.x;  a[9]=(_Float16)f2.y;  a[10]=(_Float16)f2.z; a[11]=(_Float16)f2.w;
    a[12]=(_Float16)f3.x; a[13]=(_Float16)f3.y; a[14]=(_Float16)f3.z; a[15]=(_Float16)f3.w;

    const _Float16* wr = W + (size_t)(n0 + m) * K + kc + (hs << 4);
#pragma unroll
    for (int t = 0; t < 4; ++t) {
      v16h b = ld16h(wr + (size_t)t * 16 * K);
      acc[t] = __builtin_amdgcn_wmma_f32_16x16x32_f16(
          false, a, false, b, (short)0, acc[t], false, false);
    }
  }

#pragma unroll
  for (int t = 0; t < 4; ++t) {
    int n = n0 + t * 16 + m;
    float bv = bias ? bias[n] : 0.0f;
#pragma unroll
    for (int r = 0; r < 8; ++r) {
      int row = row0 + r + (hs << 3);
      Out[(size_t)row * N + n] = actf<ACT>(acc[t][r] + bv);
    }
  }
}

// ---------------------------------------------------------------------------
// Flash attention, H=4, dh=64, S=2048. qkv: [T,768] f32 (q|k|v).
// Block = 256 thr (8 waves), block covers 128 q-rows of one (b,h); each 32-kv
// K/V chunk is staged to LDS once per block and shared by all 8 waves.
// ---------------------------------------------------------------------------
__global__ __launch_bounds__(256) void attn_kernel(const float* __restrict__ qkv,
                                                   float* __restrict__ O)
{
  __shared__ _Float16 Kl[32 * 72];        // K chunk row-major [kv][dh], stride 72
  __shared__ _Float16 Vl[64 * 40];        // V chunk transposed [dh][kv], stride 40
  __shared__ _Float16 Pl[8][16 * 40];     // per-wave P scratch [row][kv], stride 40

  int blk  = blockIdx.x;                  // 512 = B(8) * H(4) * 16
  int qb   = blk & 15;
  int h    = (blk >> 4) & 3;
  int b    = blk >> 6;
  int wv   = threadIdx.x >> 5;
  int lane = threadIdx.x & 31;
  int m = lane & 15, hs = lane >> 4, o = hs << 3;

  // Q A-fragments (softmax scale 1/8 folded in)
  int srow = qb * 128 + wv * 16 + m;
  const float* qr = qkv + ((size_t)b * SEQ + srow) * 768 + h * 64;
  v16h aq0, aq1;
#pragma unroll
  for (int j = 0; j < 8; ++j) {
    aq0[j]     = (_Float16)(qr[2 * o + j]      * 0.125f);
    aq0[j + 8] = (_Float16)(qr[2 * o + 8 + j] * 0.125f);
    aq1[j]     = (_Float16)(qr[32 + 2 * o + j] * 0.125f);
    aq1[j + 8] = (_Float16)(qr[40 + 2 * o + j] * 0.125f);
  }

  v8f oa[4];
  float mrun[8], lrun[8];
#pragma unroll
  for (int r = 0; r < 8; ++r) {
    mrun[r] = -3.0e38f; lrun[r] = 0.0f;
    oa[0][r] = 0.0f; oa[1][r] = 0.0f; oa[2][r] = 0.0f; oa[3][r] = 0.0f;
  }

  for (int c0 = 0; c0 < SEQ; c0 += 32) {
    __syncthreads();
    // K: pack two dh per b32 store; source read is a coalesced float2
    for (int e = threadIdx.x; e < 32 * 32; e += 256) {
      int r  = e >> 5;
      int d2 = (e & 31) << 1;
      const float* src = qkv + ((size_t)b * SEQ + c0 + r) * 768 + h * 64 + 256 + d2;
      float2 kf = *(const float2*)src;
      *(unsigned*)&Kl[r * 72 + d2] = pack2h(kf.x, kf.y);
    }
    // V transposed: pack two kv per b32 store
    for (int e = threadIdx.x; e < 64 * 16; e += 256) {
      int dd = e >> 4;
      int r2 = (e & 15) << 1;
      const float* src = qkv + ((size_t)b * SEQ + c0 + r2) * 768 + h * 64 + 512 + dd;
      *(unsigned*)&Vl[dd * 40 + r2] = pack2h(src[0], src[768]);
    }
    __syncthreads();

    // S = Q @ K^T : two 16x16 tiles, dh reduced in 2 WMMAs each
    v8f s0, s1;
#pragma unroll
    for (int r = 0; r < 8; ++r) { s0[r] = 0.0f; s1[r] = 0.0f; }
    s0 = __builtin_amdgcn_wmma_f32_16x16x32_f16(false, aq0, false,
           ld16h(&Kl[m * 72 + (hs << 4)]), (short)0, s0, false, false);
    s0 = __builtin_amdgcn_wmma_f32_16x16x32_f16(false, aq1, false,
           ld16h(&Kl[m * 72 + 32 + (hs << 4)]), (short)0, s0, false, false);
    s1 = __builtin_amdgcn_wmma_f32_16x16x32_f16(false, aq0, false,
           ld16h(&Kl[(16 + m) * 72 + (hs << 4)]), (short)0, s1, false, false);
    s1 = __builtin_amdgcn_wmma_f32_16x16x32_f16(false, aq1, false,
           ld16h(&Kl[(16 + m) * 72 + 32 + (hs << 4)]), (short)0, s1, false, false);

    // online softmax; row reductions via DPP16 butterflies (C-layout rows live
    // in 16-lane halves, so no cross-half movement is ever needed)
    float p0[8], p1[8];
#pragma unroll
    for (int r = 0; r < 8; ++r) {
      float t = half16_max(fmaxf(s0[r], s1[r]));
      float nm = fmaxf(mrun[r], t);
      float sc = __expf(mrun[r] - nm);
      mrun[r] = nm;
      float e0 = __expf(s0[r] - nm);
      float e1 = __expf(s1[r] - nm);
      float rs = half16_sum(e0 + e1);
      lrun[r] = lrun[r] * sc + rs;
      p0[r] = e0; p1[r] = e1;
      oa[0][r] *= sc; oa[1][r] *= sc; oa[2][r] *= sc; oa[3][r] *= sc;
    }

    // C-layout P -> A-layout via wave-private LDS round trip
    _Float16* pw = Pl[wv];
#pragma unroll
    for (int r = 0; r < 8; ++r) {
      pw[(r + (hs << 3)) * 40 + m]      = (_Float16)p0[r];
      pw[(r + (hs << 3)) * 40 + 16 + m] = (_Float16)p1[r];
    }
    // A-fragment readback: two aligned 16B LDS loads
    const v8h* prow = (const v8h*)&pw[m * 40];
    v8h plo = prow[2 * hs];        // f16 offset 2*o     (o = hs*8)
    v8h phi = prow[2 * hs + 1];    // f16 offset 2*o+8
    v16h ap;
#pragma unroll
    for (int j = 0; j < 8; ++j) { ap[j] = plo[j]; ap[j + 8] = phi[j]; }

    // O += P @ V : 4 dh tiles, kv=32 reduced in one WMMA each (V^T staged)
    oa[0] = __builtin_amdgcn_wmma_f32_16x16x32_f16(false, ap, false,
              ld16h(&Vl[m * 40 + (hs << 4)]), (short)0, oa[0], false, false);
    oa[1] = __builtin_amdgcn_wmma_f32_16x16x32_f16(false, ap, false,
              ld16h(&Vl[(16 + m) * 40 + (hs << 4)]), (short)0, oa[1], false, false);
    oa[2] = __builtin_amdgcn_wmma_f32_16x16x32_f16(false, ap, false,
              ld16h(&Vl[(32 + m) * 40 + (hs << 4)]), (short)0, oa[2], false, false);
    oa[3] = __builtin_amdgcn_wmma_f32_16x16x32_f16(false, ap, false,
              ld16h(&Vl[(48 + m) * 40 + (hs << 4)]), (short)0, oa[3], false, false);
  }

#pragma unroll
  for (int t = 0; t < 4; ++t) {
#pragma unroll
    for (int r = 0; r < 8; ++r) {
      int row = qb * 128 + wv * 16 + r + (hs << 3);
      O[((size_t)b * SEQ + row) * ED + h * 64 + t * 16 + m] = oa[t][r] / lrun[r];
    }
  }
}

// ---------------------------------------------------------------------------
// Spectral stage: per (b, quaternion group d) 2048-pt FFT x4 components in LDS,
// phase kernel, Hamilton cube on complex components, inverse FFT -> real.
// ---------------------------------------------------------------------------
struct c2 { float r, i; };
__device__ __forceinline__ c2 cmul(c2 a, c2 b) { return {a.r*b.r - a.i*b.i, a.r*b.i + a.i*b.r}; }
__device__ __forceinline__ c2 cadd(c2 a, c2 b) { return {a.r + b.r, a.i + b.i}; }
__device__ __forceinline__ c2 csub(c2 a, c2 b) { return {a.r - b.r, a.i - b.i}; }

__device__ void qmulc(const c2* a, const c2* b, c2* o) {
  o[0] = csub(csub(csub(cmul(a[0],b[0]), cmul(a[1],b[1])), cmul(a[2],b[2])), cmul(a[3],b[3]));
  o[1] = csub(cadd(cadd(cmul(a[0],b[1]), cmul(a[1],b[0])), cmul(a[2],b[3])), cmul(a[3],b[2]));
  o[2] = cadd(cadd(csub(cmul(a[0],b[2]), cmul(a[1],b[3])), cmul(a[2],b[0])), cmul(a[3],b[1]));
  o[3] = cadd(csub(cadd(cmul(a[0],b[3]), cmul(a[1],b[2])), cmul(a[2],b[1])), cmul(a[3],b[0]));
}

__device__ void fft2048(float (*re)[SEQ], float (*im)[SEQ], int tid, float sgn) {
  // bit reversal (11 bits)
  for (int idx = tid; idx < 4 * SEQ; idx += 256) {
    int c = idx >> 11;
    int s = idx & (SEQ - 1);
    int j = (int)(__brev((unsigned)s) >> 21);
    if (j > s) {
      float tr = re[c][s]; re[c][s] = re[c][j]; re[c][j] = tr;
      float ti = im[c][s]; im[c][s] = im[c][j]; im[c][j] = ti;
    }
  }
  __syncthreads();
  for (int st = 1; st <= 11; ++st) {
    int half = 1 << (st - 1);
    float ang = sgn * 3.14159265358979323846f / (float)half;
    for (int idx = tid; idx < 4 * 1024; idx += 256) {
      int c = idx >> 10;
      int k = idx & 1023;
      int pos = k & (half - 1);
      int i0 = ((k >> (st - 1)) << st) + pos;
      int i1 = i0 + half;
      float sv, cv;
      __sincosf(ang * (float)pos, &sv, &cv);
      float ur = re[c][i0], ui = im[c][i0];
      float vr = re[c][i1], vi = im[c][i1];
      float tr = vr * cv - vi * sv;
      float ti = vr * sv + vi * cv;
      re[c][i0] = ur + tr; im[c][i0] = ui + ti;
      re[c][i1] = ur - tr; im[c][i1] = ui - ti;
    }
    __syncthreads();
  }
}

__global__ __launch_bounds__(256) void spectral_kernel(const float* __restrict__ in,
                                                       float* __restrict__ out)
{
  __shared__ float re[4][SEQ];
  __shared__ float im[4][SEQ];
  int b = blockIdx.x >> 6;
  int d = blockIdx.x & 63;
  int tid = threadIdx.x;
  size_t base = (size_t)b * SEQ * ED + (size_t)d * 4;
  for (int c = 0; c < 4; ++c)
    for (int s = tid; s < SEQ; s += 256) {
      re[c][s] = in[base + (size_t)s * ED + c];
      im[c][s] = 0.0f;
    }
  __syncthreads();
  fft2048(re, im, tid, -1.0f);                  // forward (e^{-i2πkn/N})
  for (int k = tid; k < SEQ; k += 256) {
    float th = 1.5f * atanf(logf((float)k + 1e-10f));
    float cr = cosf(th), ci = sinf(th);
    c2 f[4], q2[4], q3[4];
#pragma unroll
    for (int c = 0; c < 4; ++c) {
      c2 v = {re[c][k], im[c][k]};
      f[c] = cmul(v, {cr, ci});
    }
    qmulc(f, f, q2);
    qmulc(q2, f, q3);
#pragma unroll
    for (int c = 0; c < 4; ++c) { re[c][k] = q3[c].r; im[c][k] = q3[c].i; }
  }
  __syncthreads();
  fft2048(re, im, tid, 1.0f);                   // inverse (unscaled)
  const float inv = 1.0f / (float)SEQ;
  for (int c = 0; c < 4; ++c)
    for (int s = tid; s < SEQ; s += 256)
      out[base + (size_t)s * ED + c] = re[c][s] * inv;
}

// ---------------------------------------------------------------------------
// Fused element-wise / reduction kernels (wave-per-token where a row dot or
// row variance is needed).
// ---------------------------------------------------------------------------
__global__ void stab_kernel(const float* __restrict__ x, float* __restrict__ stab) {
  int b = blockIdx.x;
  int lane = threadIdx.x & 31;
  int w = threadIdx.x >> 5;
  float s1 = 0.0f, s2 = 0.0f;
  for (int s = w; s < SEQ; s += 8) {
    const float* row = x + ((size_t)b * SEQ + s) * ED;
    float ss = 0.0f;
#pragma unroll
    for (int j = 0; j < 8; ++j) { float v = row[lane * 8 + j]; ss += v * v; }
    ss = wave_sum(ss);
    if (lane == 0) { s1 += sqrtf(ss); s2 += ss; }
  }
  __shared__ float a1[8], a2[8];
  if (lane == 0) { a1[w] = s1; a2[w] = s2; }
  __syncthreads();
  if (threadIdx.x == 0) {
    float t1 = 0.0f, t2 = 0.0f;
    for (int i = 0; i < 8; ++i) { t1 += a1[i]; t2 += a2[i]; }
    float n = (float)SEQ;
    float var = (t2 - t1 * t1 / n) / (n - 1.0f);   // ddof=1
    stab[b] = 1.0f / (1.0f + var);
  }
}

__global__ void dopa_apply_kernel(const float* __restrict__ x, const float* __restrict__ H,
                                  const float* __restrict__ w2, const float* __restrict__ b2,
                                  const float* __restrict__ stab, float* __restrict__ X1) {
  int t = blockIdx.x * 8 + (threadIdx.x >> 5);
  int lane = threadIdx.x & 31;
  const float* hr = H + (size_t)t * 128;
  float dd = 0.0f;
#pragma unroll
  for (int j = 0; j < 4; ++j) dd += hr[lane * 4 + j] * w2[lane * 4 + j];
  dd = wave_sum(dd);
  float resp = 1.0f / (1.0f + __expf(-(dd + b2[0])));
  float g = 1.0f + 0.8f * resp * stab[t >> 11];
#pragma unroll
  for (int j = 0; j < 8; ++j) {
    size_t i = (size_t)t * ED + lane * 8 + j;
    X1[i] = x[i] * g;
  }
}

__global__ void ln_kernel(const float* __restrict__ X, const float* __restrict__ g,
                          const float* __restrict__ bb, float* __restrict__ XN) {
  int t = blockIdx.x * 8 + (threadIdx.x >> 5);
  int lane = threadIdx.x & 31;
  const float* xr = X + (size_t)t * ED;
  float s1 = 0.0f, s2 = 0.0f;
#pragma unroll
  for (int j = 0; j < 8; ++j) { float v = xr[lane * 8 + j]; s1 += v; s2 += v * v; }
  s1 = wave_sum(s1); s2 = wave_sum(s2);
  float mu = s1 * (1.0f / 256.0f);
  float var = s2 * (1.0f / 256.0f) - mu * mu;      // ddof=0
  float rs = rsqrtf(var + 1e-5f);
#pragma unroll
  for (int j = 0; j < 8; ++j) {
    int c = lane * 8 + j;
    XN[(size_t)t * ED + c] = (xr[c] - mu) * rs * g[c] + bb[c];
  }
}

__global__ void sero_combine_kernel(float* __restrict__ X1, const float* __restrict__ S1,
                                    const float* __restrict__ S2, const float* __restrict__ HARM) {
  size_t i = (size_t)blockIdx.x * 256 + threadIdx.x;
  X1[i] = (0.6f * X1[i] + 0.4f * HARM[i]) * (S1[i] * S2[i]);
}

__global__ void chol_combine_kernel(const float* __restrict__ NIC, const float* __restrict__ MUS,
                                    float* __restrict__ CH) {
  size_t i = (size_t)blockIdx.x * 256 + threadIdx.x;
  CH[i] = 2.0f * (0.9f * NIC[i] + 0.1f * MUS[i]);   // ACH_SEL * (focus mix)
}

__global__ void mul_kernel(float* __restrict__ X, const float* __restrict__ A) {
  size_t i = (size_t)blockIdx.x * 256 + threadIdx.x;
  X[i] = X[i] * A[i];
}

__global__ void gaba_kernel(float* __restrict__ X, const float* __restrict__ GAH,
                            const float* __restrict__ gw2, const float* __restrict__ gb2,
                            const float* __restrict__ thr, const float* __restrict__ GB) {
  int t = blockIdx.x * 8 + (threadIdx.x >> 5);
  int lane = threadIdx.x & 31;
  const float* xr = X + (size_t)t * ED;
  float s1 = 0.0f, s2 = 0.0f;
#pragma unroll
  for (int j = 0; j < 8; ++j) { float v = xr[lane * 8 + j]; s1 += v; s2 += v * v; }
  s1 = wave_sum(s1); s2 = wave_sum(s2);
  float noise = (s2 - s1 * s1 * (1.0f / 256.0f)) * (1.0f / 255.0f);  // ddof=1
  const float* hr = GAH + (size_t)t * 64;
  float dd = hr[lane * 2] * gw2[lane * 2] + hr[lane * 2 + 1] * gw2[lane * 2 + 1];
  dd = wave_sum(dd);
  float ga = 1.0f / (1.0f + __expf(-(dd + gb2[0])));
  float inh = (noise > thr[0]) ? 0.4f * ga : 0.0f;
  float f = 1.0f - inh;
#pragma unroll
  for (int j = 0; j < 8; ++j) {
    size_t i = (size_t)t * ED + lane * 8 + j;
    X[i] = (X[i] + GB[i]) * f;
  }
}

__global__ void glut_kernel(const float* __restrict__ X, const float* __restrict__ AM,
                            const float* __restrict__ NM, const float* __restrict__ KA,
                            const float* __restrict__ gw, const float* __restrict__ gbias,
                            float* __restrict__ out) {
  int t = blockIdx.x * 8 + (threadIdx.x >> 5);
  int lane = threadIdx.x & 31;
  const float* xr = X + (size_t)t * ED;
  float dd = 0.0f;
#pragma unroll
  for (int j = 0; j < 8; ++j) dd += xr[lane * 8 + j] * gw[lane * 8 + j];
  dd = wave_sum(dd);
  float gate = 1.0f / (1.0f + __expf(-(dd + gbias[0])));
#pragma unroll
  for (int j = 0; j < 8; ++j) {
    size_t i = (size_t)t * ED + lane * 8 + j;
    out[i] = X[i] + 1.1f * gate * (0.5f * AM[i] + 0.3f * NM[i] + 0.2f * KA[i]);
  }
}

// ---------------------------------------------------------------------------
// Host orchestration
// ---------------------------------------------------------------------------
extern "C" void kernel_launch(void* const* d_in, const int* in_sizes, int n_in,
                              void* d_out, int out_size, void* d_ws, size_t ws_size,
                              hipStream_t stream)
{
  (void)in_sizes; (void)n_in; (void)out_size; (void)ws_size;

  const float* x       = (const float*)d_in[0];
  const float* da_w1   = (const float*)d_in[1];
  const float* da_b1   = (const float*)d_in[2];
  const float* da_w2   = (const float*)d_in[3];
  const float* da_b2   = (const float*)d_in[4];
  const float* ht1_w   = (const float*)d_in[5];
  const float* ht1_b   = (const float*)d_in[6];
  const float* ht2_w   = (const float*)d_in[7];
  const float* ht2_b   = (const float*)d_in[8];
  const float* ln_g    = (const float*)d_in[9];
  const float* ln_b    = (const float*)d_in[10];
  const float* harm_w  = (const float*)d_in[11];
  const float* harm_b  = (const float*)d_in[12];
  const float* mha_in_w  = (const float*)d_in[13];
  const float* mha_in_b  = (const float*)d_in[14];
  const float* mha_out_w = (const float*)d_in[15];
  const float* mha_out_b = (const float*)d_in[16];
  const float* mus_w1  = (const float*)d_in[17];
  const float* mus_b1  = (const float*)d_in[18];
  const float* mus_w2  = (const float*)d_in[19];
  const float* mus_b2  = (const float*)d_in[20];
  const float* ga_w1   = (const float*)d_in[21];
  const float* ga_b1   = (const float*)d_in[22];
  const float* ga_w2   = (const float*)d_in[23];
  const float* ga_b2   = (const float*)d_in[24];
  const float* gb_w    = (const float*)d_in[25];
  const float* gb_b    = (const float*)d_in[26];
  const float* noise_thr = (const float*)d_in[27];
  const float* ampa_w  = (const float*)d_in[28];
  const float* ampa_b  = (const float*)d_in[29];
  const float* nmda_w  = (const float*)d_in[30];
  const float* nmda_b  = (const float*)d_in[31];
  const float* kai_w   = (const float*)d_in[32];
  const float* kai_b   = (const float*)d_in[33];
  const float* gate_w  = (const float*)d_in[34];
  const float* gate_b  = (const float*)d_in[35];

  char* ws = (char*)d_ws;
  _Float16* f_daw1   = (_Float16*)ws;          // 128*256
  _Float16* f_ht1    = f_daw1  + 128 * 256;
  _Float16* f_ht2    = f_ht1   + 256 * 256;
  _Float16* f_harm   = f_ht2   + 256 * 256;
  _Float16* f_mhain  = f_harm  + 256 * 256;    // 768*256
  _Float16* f_mhaout = f_mhain + 768 * 256;
  _Float16* f_mus1   = f_mhaout+ 256 * 256;
  _Float16* f_mus2   = f_mus1  + 256 * 256;
  _Float16* f_gaw1   = f_mus2  + 256 * 256;    // 64*256
  _Float16* f_gb     = f_gaw1  + 64 * 256;
  _Float16* f_ampa   = f_gb    + 256 * 256;
  _Float16* f_nmda   = f_ampa  + 256 * 256;
  _Float16* f_kai    = f_nmda  + 256 * 256;

  float* stab = (float*)(ws + ((size_t)2 << 20));
  float* X1   = (float*)(ws + ((size_t)4 << 20));
  float* T1   = X1 + (size_t)TOK * ED;
  float* T2   = T1 + (size_t)TOK * ED;
  float* T3   = T2 + (size_t)TOK * ED;
  float* T4   = T3 + (size_t)TOK * ED;
  float* QKV  = T4 + (size_t)TOK * ED;         // [T,768]
  float* outp = (float*)d_out;

  auto cvt = [&](const float* s, _Float16* d, int n) {
    cvt_f16_kernel<<<dim3((unsigned)((n + 255) / 256)), dim3(256), 0, stream>>>(s, d, n);
  };
  cvt(da_w1,     f_daw1,   128 * 256);
  cvt(ht1_w,     f_ht1,    256 * 256);
  cvt(ht2_w,     f_ht2,    256 * 256);
  cvt(harm_w,    f_harm,   256 * 256);
  cvt(mha_in_w,  f_mhain,  768 * 256);
  cvt(mha_out_w, f_mhaout, 256 * 256);
  cvt(mus_w1,    f_mus1,   256 * 256);
  cvt(mus_w2,    f_mus2,   256 * 256);
  cvt(ga_w1,     f_gaw1,   64 * 256);
  cvt(gb_w,      f_gb,     256 * 256);
  cvt(ampa_w,    f_ampa,   256 * 256);
  cvt(nmda_w,    f_nmda,   256 * 256);
  cvt(kai_w,     f_kai,    256 * 256);

  auto gemm = [&](int act, const float* X, const _Float16* W, const float* B, float* O, int N) {
    dim3 g((unsigned)(((TOK / 16) * (N / 64)) / 8));
    switch (act) {
      case 0: gemm_kernel<0><<<g, 256, 0, stream>>>(X, W, B, O, N); break;
      case 1: gemm_kernel<1><<<g, 256, 0, stream>>>(X, W, B, O, N); break;
      case 2: gemm_kernel<2><<<g, 256, 0, stream>>>(X, W, B, O, N); break;
      case 3: gemm_kernel<3><<<g, 256, 0, stream>>>(X, W, B, O, N); break;
      case 4: gemm_kernel<4><<<g, 256, 0, stream>>>(X, W, B, O, N); break;
    }
  };

  const unsigned FLAT = (unsigned)((size_t)TOK * ED / 256);   // 16384 blocks
  const unsigned TOKB = (unsigned)(TOK / 8);                  // wave-per-token blocks

  // --- dopamine ---
  stab_kernel<<<dim3(NB), dim3(256), 0, stream>>>(x, stab);
  gemm(1, x, f_daw1, da_b1, T1, 128);                          // h = relu(x W1^T + b1)
  dopa_apply_kernel<<<dim3(TOKB), dim3(256), 0, stream>>>(x, T1, da_w2, da_b2, stab, X1);

  // --- serotonin ---
  gemm(2, X1, f_ht1, ht1_b, T1, 256);                          // s1
  gemm(3, X1, f_ht2, ht2_b, T2, 256);                          // s2
  ln_kernel<<<dim3(TOKB), dim3(256), 0, stream>>>(X1, ln_g, ln_b, T3);
  gemm(3, T3, f_harm, harm_b, T4, 256);                        // harm
  sero_combine_kernel<<<dim3(FLAT), dim3(256), 0, stream>>>(X1, T1, T2, T4);

  // --- acetylcholine ---
  gemm(0, X1, f_mhain, mha_in_b, QKV, 768);
  attn_kernel<<<dim3(512), dim3(256), 0, stream>>>(QKV, T1);
  gemm(0, T1, f_mhaout, mha_out_b, T2, 256);                   // nic
  gemm(4, X1, f_mus1, mus_b1, T3, 256);                        // gelu
  gemm(0, T3, f_mus2, mus_b2, T4, 256);                        // mus
  chol_combine_kernel<<<dim3(FLAT), dim3(256), 0, stream>>>(T2, T4, T3);
  spectral_kernel<<<dim3(NB * 64), dim3(256), 0, stream>>>(T3, T4);
  mul_kernel<<<dim3(FLAT), dim3(256), 0, stream>>>(X1, T4);

  // --- GABA ---
  gemm(1, X1, f_gaw1, ga_b1, T1, 64);
  gemm(3, X1, f_gb, gb_b, T2, 256);
  gaba_kernel<<<dim3(TOKB), dim3(256), 0, stream>>>(X1, T1, ga_w2, ga_b2, noise_thr, T2);

  // --- glutamate ---
  gemm(1, X1, f_ampa, ampa_b, T1, 256);
  gemm(2, X1, f_nmda, nmda_b, T2, 256);
  gemm(3, X1, f_kai, kai_b, T3, 256);
  glut_kernel<<<dim3(TOKB), dim3(256), 0, stream>>>(X1, T1, T2, T3, gate_w, gate_b, outp);
}
